// NeuronQwen3NextGatedDeltaNet_23948737642627
// MI455X (gfx1250) — matrix operand order repaired and checked by reading, not compiled, not run on hardware
//
#include <hip/hip_runtime.h>
#include <hip/hip_bf16.h>
#include <math.h>

#define H_K 16
#define H_V 32
#define DK 128
#define DV 128
#define HIDDEN 2048
#define KEY_DIM 2048
#define VALUE_DIM 4096
#define CONV_DIM 8192
#define QKVZ_DIM 12288
#define SEQ 2048
#define BATCH 2
#define ROWS (BATCH * SEQ)   /* 4096 */
#define CHUNKLEN 64
#define NCHUNK (SEQ / CHUNKLEN)

/* GEMM macro-tile */
#define GM 128
#define GN 64
#define GK 32

typedef __bf16 bf16_t;
typedef bf16_t v16bf __attribute__((ext_vector_type(16)));
typedef float  v8f   __attribute__((ext_vector_type(8)));
typedef float  v2f   __attribute__((ext_vector_type(2)));
typedef unsigned int v4u __attribute__((ext_vector_type(4)));
typedef int v8i __attribute__((ext_vector_type(8)));
typedef int v4i __attribute__((ext_vector_type(4)));

#if __has_builtin(__builtin_amdgcn_tensor_load_to_lds) && \
    __has_builtin(__builtin_amdgcn_s_wait_tensorcnt)
#define HAVE_TDM 1
#else
#define HAVE_TDM 0
#endif

#if HAVE_TDM
// ---------------------------------------------------------------------------
// TDM 2D tile load: rows x cols f32 tile (row stride = stride_elems) from
// global into LDS at byte offset lds_addr. D# per CDNA5 ISA ch.8:
//   group0: count=1 | lds_addr | global_addr[56:0] | type=2
//   group1: data_size=4B | tensor_dim0/1 | tile_dim0/1 | dim0 stride
// 6-arg builtin (clang-23 / therock headers): (g0, g1, g2, g3, g_extra, cpol).
// ---------------------------------------------------------------------------
__device__ inline void tdm_load_2d(const float* gsrc, unsigned lds_addr,
                                   unsigned rows, unsigned cols,
                                   unsigned long long stride_elems) {
  const unsigned long long ga = (unsigned long long)(size_t)gsrc;
  v4u g0;
  g0[0] = 1u;                                               // count=1, user
  g0[1] = lds_addr;                                         // LDS byte addr
  g0[2] = (unsigned)ga;                                     // global[31:0]
  g0[3] = (unsigned)((ga >> 32) & 0x01FFFFFFu) | (2u << 30);// global hi|type=2
  v8i g1;
  g1[0] = (int)(2u << 16);                                  // data_size=4B
  g1[1] = (int)((cols & 0xFFFFu) << 16);                    // tensor_dim0 lo
  g1[2] = (int)((cols >> 16) | ((rows & 0xFFFFu) << 16));   // d0 hi|d1 lo
  g1[3] = (int)((rows >> 16) | ((cols & 0xFFFFu) << 16));   // d1 hi|tile_dim0
  g1[4] = (int)(rows & 0xFFFFu);                            // tile_dim1, td2=0
  g1[5] = (int)(unsigned)stride_elems;                      // stride0[31:0]
  g1[6] = (int)((unsigned)(stride_elems >> 32) & 0xFFFFu);  // stride0 hi
  g1[7] = 0;                                                // stride1=0 (2D)
  v4i z4 = {0, 0, 0, 0};
  v8i z8 = {0, 0, 0, 0, 0, 0, 0, 0};
  __builtin_amdgcn_tensor_load_to_lds(g0, g1, z4, z4, z8, 0);
}
#endif

// ---------------------------------------------------------------------------
// f32 matrix-pipe primitive: D = A(16x4) x B(4x16) + C, full f32 precision.
// ---------------------------------------------------------------------------
__device__ inline v8f wmma4_f32(v2f a, v2f b, v8f c) {
#if __has_builtin(__builtin_amdgcn_wmma_f32_16x16x4_f32)
  return __builtin_amdgcn_wmma_f32_16x16x4_f32(false, a, false, b, (short)0, c,
                                               false, false);
#else
  c[0] += a[0] * b[0] + a[1] * b[1];  // compile-safe fallback
  return c;
#endif
}

// Generic LDS->WMMA f32 matmul over (Mt*16) x (Nt*16) output tiles, K-dim K.
// ISA 7.12.2 layouts: A 16x4: a0=K(2*hi), a1=K(2*hi+1), m=lane&15.
//                     B 4x16: b0=B[2*hi][n], b1=B[2*hi+1][n], n=lane&15.
//                     C/D: elem r -> M = 16*mt + 8*hi + r, N = 16*nt + lo.
template <typename FA, typename FB, typename FC>
__device__ inline void wmma_mm(int Mt, int Nt, int K, int lane, int wave,
                               FA fa, FB fb, FC fc) {
  const int lo = lane & 15;
  const int hi = lane >> 4;
  const int ntile = Mt * Nt;
  for (int t = wave; t < ntile; t += 8) {
    const int mt = t / Nt, nt = t % Nt;
    const int m = mt * 16 + lo;
    const int n = nt * 16 + lo;
    v8f acc = {};
    for (int k0 = 0; k0 < K; k0 += 4) {
      v2f a, b;
      a[0] = fa(m, k0 + 2 * hi);
      a[1] = fa(m, k0 + 2 * hi + 1);
      b[0] = fb(k0 + 2 * hi, n);
      b[1] = fb(k0 + 2 * hi + 1, n);
      acc = wmma4_f32(a, b, acc);
    }
#pragma unroll
    for (int r = 0; r < 8; ++r) fc(mt * 16 + hi * 8 + r, n, acc[r]);
  }
}

// ---------------------------------------------------------------------------
// GEMM: out[M,N] = A[M,K] * W[N,K]^T  (f32 mem, bf16 WMMA compute)
// 256 thr = 8 waves, 128x64 tile. TDM double-buffers f32 K-slices into LDS;
// operands converted f32->bf16 in the gather (co-issues with WMMA).
// ---------------------------------------------------------------------------
__global__ __launch_bounds__(256) void gemm_bf16_wmma(
    const float* __restrict__ A, const float* __restrict__ W,
    float* __restrict__ out, int M, int N, int K) {
  (void)M;
  extern __shared__ float gsm[];
  float* sA = gsm;                 // 2 stages of GM*GK f32
  float* sB = gsm + 2 * GM * GK;   // 2 stages of GN*GK f32
  const int m0 = blockIdx.y * GM;
  const int n0 = blockIdx.x * GN;
  const int tid = threadIdx.x;
  const int lane = tid & 31;
  const int wave = tid >> 5;
  const int lo = lane & 15;
  const int hi = lane >> 4;
  const int nt = wave & 3;        // waves w and w+4 share an N-tile column
  const int mtb = wave >> 2;      // 0/1 -> mt = mtb + 2*i
  v8f acc[4] = {{}, {}, {}, {}};

  auto issue_stage = [&](int k0, int stg) {
#if HAVE_TDM
    if (wave == 0) {
      tdm_load_2d(A + (size_t)m0 * K + k0,
                  (unsigned)(size_t)(sA + stg * GM * GK), GM, GK,
                  (unsigned long long)K);
      tdm_load_2d(W + (size_t)n0 * K + k0,
                  (unsigned)(size_t)(sB + stg * GN * GK), GN, GK,
                  (unsigned long long)K);
    }
#else
    for (int t = tid; t < GM * GK; t += 256)
      sA[stg * GM * GK + t] = A[(size_t)(m0 + (t >> 5)) * K + k0 + (t & 31)];
    for (int t = tid; t < GN * GK; t += 256)
      sB[stg * GN * GK + t] = W[(size_t)(n0 + (t >> 5)) * K + k0 + (t & 31)];
#endif
  };

  issue_stage(0, 0);
#if HAVE_TDM
  if (wave == 0) __builtin_amdgcn_s_wait_tensorcnt(0);
#endif
  __syncthreads();

  for (int k0 = 0; k0 < K; k0 += GK) {
    const int cur = (k0 / GK) & 1;
    if (k0 + GK < K) issue_stage(k0 + GK, cur ^ 1);
    const float* cA = sA + cur * GM * GK;
    const float* cB = sB + cur * GN * GK;
    v16bf bvec;
    {
      const float* pb = cB + (nt * 16 + lo) * GK + hi * 16;
#pragma unroll
      for (int e = 0; e < 16; ++e) bvec[e] = (bf16_t)pb[e];
    }
#pragma unroll
    for (int i = 0; i < 4; ++i) {
      const int mt = mtb + 2 * i;
      const float* pa = cA + (mt * 16 + lo) * GK + hi * 16;
      v16bf avec;
#pragma unroll
      for (int e = 0; e < 8; ++e) {
        avec[e] = (bf16_t)pa[e];
        avec[8 + e] = (bf16_t)pa[8 + e];
      }
      acc[i] = __builtin_amdgcn_wmma_f32_16x16x32_bf16(
          false, avec, false, bvec, (short)0, acc[i], false, false);
    }
#if HAVE_TDM
    if (wave == 0) __builtin_amdgcn_s_wait_tensorcnt(0);
#endif
    __syncthreads();
  }
#pragma unroll
  for (int i = 0; i < 4; ++i) {
    const int mt = mtb + 2 * i;
    const int n = n0 + nt * 16 + lo;
#pragma unroll
    for (int rr = 0; rr < 8; ++rr)
      out[(size_t)(m0 + mt * 16 + hi * 8 + rr) * N + n] = acc[i][rr];
  }
}

// ---------------------------------------------------------------------------
// depthwise causal conv(K=4) + SiLU over first 8192 channels.
// ---------------------------------------------------------------------------
__global__ __launch_bounds__(256) void conv_silu(
    const float* __restrict__ qkvz, const float* __restrict__ w,
    float* __restrict__ mixed) {
  const size_t idx = (size_t)blockIdx.x * 256 + threadIdx.x;
  const int ch = (int)(idx & (CONV_DIM - 1));
  const size_t rs = idx >> 13;            // (b*SEQ + s)
  const int s = (int)(rs & (SEQ - 1));
  float acc = 0.f;
#pragma unroll
  for (int j = 0; j < 4; ++j) {
    const int sp = s - 3 + j;
    if (sp >= 0)
      acc += w[ch * 4 + j] * qkvz[(rs - (size_t)(s - sp)) * QKVZ_DIM + ch];
  }
  mixed[idx] = acc / (1.f + __expf(-acc));
}

// ---------------------------------------------------------------------------
// beta = sigmoid(b), g = -exp(A_log)*softplus(a + dt_bias)
// ---------------------------------------------------------------------------
__global__ __launch_bounds__(256) void gates_kernel(
    const float* __restrict__ ba, const float* __restrict__ A_log,
    const float* __restrict__ dt_bias, float* __restrict__ beta,
    float* __restrict__ g) {
  const int idx = blockIdx.x * 256 + threadIdx.x;  // ROWS*H_V
  const int h = idx & (H_V - 1);
  const int rs = idx >> 5;
  const float bv = ba[(size_t)rs * 64 + h];
  const float av = ba[(size_t)rs * 64 + H_V + h] + dt_bias[h];
  beta[idx] = 1.f / (1.f + __expf(-bv));
  const float sp = (av > 30.f) ? av : log1pf(__expf(av));
  g[idx] = -__expf(A_log[h]) * sp;
}

// ---------------------------------------------------------------------------
// in-place L2 norm of q (plus dk^-0.5) and k rows. Wave per row.
// ---------------------------------------------------------------------------
__global__ __launch_bounds__(256) void l2norm_qk(float* __restrict__ mixed) {
  const int rid = blockIdx.x * 8 + (threadIdx.x >> 5);  // ROWS*32 rows
  const int lane = threadIdx.x & 31;
  const int rs = rid >> 5;
  const int j = rid & 31;
  float* p = mixed + (size_t)rs * CONV_DIM +
             ((j < 16) ? j * DK : KEY_DIM + (j - 16) * DK);
  float4 x = *(const float4*)(p + lane * 4);
  float ss = x.x * x.x + x.y * x.y + x.z * x.z + x.w * x.w;
#pragma unroll
  for (int m = 16; m >= 1; m >>= 1) ss += __shfl_xor(ss, m, 32);
  float inv = 1.f / (sqrtf(ss) + 1e-6f);
  if (j < 16) inv *= 0.08838834764831845f;  // dk^-0.5
  x.x *= inv; x.y *= inv; x.z *= inv; x.w *= inv;
  *(float4*)(p + lane * 4) = x;
}

// ---------------------------------------------------------------------------
// chunked gated delta rule. One block per (b,h); 8 waves; state + chunk
// tiles in LDS (~257KB of the 320KB/WGP). All matmuls on the f32 WMMA pipe.
// ---------------------------------------------------------------------------
__global__ __launch_bounds__(256) void gdn_core(
    const float* __restrict__ mixed, const float* __restrict__ beta,
    const float* __restrict__ g, float* __restrict__ outc) {
  extern __shared__ float smem[];
  float* st = smem;                //128*128 running state
  float* kc = st + 128 * 128;      // 64*128 k chunk (l2-normed)
  float* qc = kc + 64 * 128;       // 64*128 q chunk (scaled)
  float* vb = qc + 64 * 128;       // 64*128 v*beta, later v_new
  float* val = vb + 64 * 128;      // 64*128 T@v_beta, later attn_inter
  float* kcd = val + 64 * 128;     // 64*128 k_cumdecay
  float* Am = kcd + 64 * 128;      // 64*64 A, later attn
  float* Tm = Am + 64 * 64;        // 64*64 (I-A)^-1
  float* gcs = Tm + 64 * 64;       // 64 cumsum(g)
  float* bet = gcs + 64;           // 64 beta

  const int b = blockIdx.x >> 5;
  const int h = blockIdx.x & 31;
  const int tid = threadIdx.x;
  const int lane = tid & 31;
  const int wave = tid >> 5;

  for (int t = tid; t < 128 * 128; t += 256) st[t] = 0.f;
  __syncthreads();

  for (int cix = 0; cix < NCHUNK; ++cix) {
    const int s0 = cix * CHUNKLEN;
    if (tid < 64) {
      const size_t rs = (size_t)(b * SEQ + s0 + tid);
      bet[tid] = beta[rs * H_V + h];
      gcs[tid] = g[rs * H_V + h];
    }
    __syncthreads();
    if (tid == 0) {
      float a = 0.f;
      for (int i = 0; i < CHUNKLEN; ++i) { a += gcs[i]; gcs[i] = a; }
    }
    __syncthreads();
    for (int t = tid; t < 64 * 128; t += 256) {
      const int i = t >> 7, d = t & 127;
      const size_t rowm = (size_t)(b * SEQ + s0 + i) * CONV_DIM;
      qc[t] = mixed[rowm + (h >> 1) * DK + d];
      kc[t] = mixed[rowm + KEY_DIM + (h >> 1) * DK + d];
      vb[t] = mixed[rowm + 2 * KEY_DIM + h * DV + d] * bet[i];
    }
    __syncthreads();
    // A = -(Kb x K^T) .* decay (strictly lower)
    wmma_mm(4, 4, 128, lane, wave,
            [&](int m, int k) { return kc[m * 128 + k] * bet[m]; },
            [&](int k, int n) { return kc[n * 128 + k]; },
            [&](int m, int n, float v) {
              Am[m * 64 + n] = (n < m) ? -v * __expf(gcs[m] - gcs[n]) : 0.f;
            });
    __syncthreads();
    for (int t = tid; t < 64 * 64; t += 256) Tm[t] = 0.f;
    __syncthreads();
    if (tid < 64) {  // T = (I - A)^-1, column `tid`: no cross-thread deps
      const int col = tid;
      for (int i = col; i < 64; ++i) {
        float s = (i == col) ? 1.f : 0.f;
        for (int j = col; j < i; ++j) s += Am[i * 64 + j] * Tm[j * 64 + col];
        Tm[i * 64 + col] = s;
      }
    }
    __syncthreads();
    // value_ = T @ v_beta ; k_cumdecay = T @ (k_beta * e^gc)
    wmma_mm(4, 8, 64, lane, wave,
            [&](int m, int k) { return Tm[m * 64 + k]; },
            [&](int k, int n) { return vb[k * 128 + n]; },
            [&](int m, int n, float v) { val[m * 128 + n] = v; });
    wmma_mm(4, 8, 64, lane, wave,
            [&](int m, int k) { return Tm[m * 64 + k]; },
            [&](int k, int n) {
              return kc[k * 128 + n] * bet[k] * __expf(gcs[k]);
            },
            [&](int m, int n, float v) { kcd[m * 128 + n] = v; });
    __syncthreads();
    // v_new = value_ - kcd @ state (into vb) ; attn = (Q K^T) .* decay (tril)
    wmma_mm(4, 8, 128, lane, wave,
            [&](int m, int k) { return kcd[m * 128 + k]; },
            [&](int k, int n) { return st[k * 128 + n]; },
            [&](int m, int n, float v) { vb[m * 128 + n] = val[m * 128 + n] - v; });
    wmma_mm(4, 4, 128, lane, wave,
            [&](int m, int k) { return qc[m * 128 + k]; },
            [&](int k, int n) { return kc[n * 128 + k]; },
            [&](int m, int n, float v) {
              Am[m * 64 + n] = (n <= m) ? v * __expf(gcs[m] - gcs[n]) : 0.f;
            });
    __syncthreads();
    // attn_inter = (q * e^gc) @ state (into val)
    wmma_mm(4, 8, 128, lane, wave,
            [&](int m, int k) { return qc[m * 128 + k] * __expf(gcs[m]); },
            [&](int k, int n) { return st[k * 128 + n]; },
            [&](int m, int n, float v) { val[m * 128 + n] = v; });
    __syncthreads();
    // out = attn_inter + attn @ v_new
    wmma_mm(4, 8, 64, lane, wave,
            [&](int m, int k) { return Am[m * 64 + k]; },
            [&](int k, int n) { return vb[k * 128 + n]; },
            [&](int m, int n, float v) {
              outc[((size_t)(b * SEQ + s0 + m) * H_V + h) * DV + n] =
                  val[m * 128 + n] + v;
            });
    __syncthreads();
    // state = state*e^g_last + (k * e^(g_last-gc))^T @ v_new
    const float gl = gcs[63];
    const float egl = __expf(gl);
    wmma_mm(8, 8, 64, lane, wave,
            [&](int m, int k) { return kc[k * 128 + m] * __expf(gl - gcs[k]); },
            [&](int k, int n) { return vb[k * 128 + n]; },
            [&](int m, int n, float v) {
              st[m * 128 + n] = st[m * 128 + n] * egl + v;
            });
    __syncthreads();
  }
}

// ---------------------------------------------------------------------------
// in-place RMSNorm over DV, gate with norm_weight * SiLU(z).
// ---------------------------------------------------------------------------
__global__ __launch_bounds__(256) void rmsnorm_gate(
    float* __restrict__ core, const float* __restrict__ qkvz,
    const float* __restrict__ nw) {
  const int rid = blockIdx.x * 8 + (threadIdx.x >> 5);  // ROWS*H_V rows
  const int lane = threadIdx.x & 31;
  const int rs = rid >> 5;
  const int h = rid & 31;
  float* p = core + (size_t)rid * DV;
  const float* pz = qkvz + (size_t)rs * QKVZ_DIM + (2 * KEY_DIM + VALUE_DIM) +
                    h * DV;
  float4 x = *(const float4*)(p + lane * 4);
  float ss = x.x * x.x + x.y * x.y + x.z * x.z + x.w * x.w;
#pragma unroll
  for (int m = 16; m >= 1; m >>= 1) ss += __shfl_xor(ss, m, 32);
  const float inv = rsqrtf(ss * (1.f / 128.f) + 1e-6f);
  float4 z4 = *(const float4*)(pz + lane * 4);
  float4 w4 = *(const float4*)(nw + lane * 4);
  x.x = w4.x * (x.x * inv) * (z4.x / (1.f + __expf(-z4.x)));
  x.y = w4.y * (x.y * inv) * (z4.y / (1.f + __expf(-z4.y)));
  x.z = w4.z * (x.z * inv) * (z4.z / (1.f + __expf(-z4.z)));
  x.w = w4.w * (x.w * inv) * (z4.w / (1.f + __expf(-z4.w)));
  *(float4*)(p + lane * 4) = x;
}

// ---------------------------------------------------------------------------
extern "C" void kernel_launch(void* const* d_in, const int* in_sizes, int n_in,
                              void* d_out, int out_size, void* d_ws,
                              size_t ws_size, hipStream_t stream) {
  (void)in_sizes; (void)n_in; (void)out_size; (void)ws_size;
  const float* X   = (const float*)d_in[0];  // hidden_states (2,2048,2048)
  const float* Wq  = (const float*)d_in[1];  // W_qkvz (12288,2048)
  const float* Wba = (const float*)d_in[2];  // W_ba (64,2048)
  const float* cw  = (const float*)d_in[3];  // conv_w (8192,1,4)
  const float* dtb = (const float*)d_in[4];  // dt_bias (32)
  const float* Alg = (const float*)d_in[5];  // A_log (32)
  const float* nw  = (const float*)d_in[6];  // norm_weight (128)
  const float* Wo  = (const float*)d_in[7];  // W_out (2048,4096)
  float* out = (float*)d_out;

  float* ws    = (float*)d_ws;
  float* qkvz  = ws;                                    // 4096*12288
  float* ba    = qkvz  + (size_t)ROWS * QKVZ_DIM;       // 4096*64
  float* mixed = ba    + (size_t)ROWS * 64;             // 4096*8192
  float* beta  = mixed + (size_t)ROWS * CONV_DIM;       // 4096*32
  float* g     = beta  + (size_t)ROWS * H_V;            // 4096*32
  float* core  = g     + (size_t)ROWS * H_V;            // 4096*32*128

  const size_t gemm_lds = (size_t)(2 * (GM + GN) * GK) * sizeof(float); // 48KB
  gemm_bf16_wmma<<<dim3(QKVZ_DIM / GN, ROWS / GM), 256, gemm_lds, stream>>>(
      X, Wq, qkvz, ROWS, QKVZ_DIM, HIDDEN);
  gemm_bf16_wmma<<<dim3(64 / GN, ROWS / GM), 256, gemm_lds, stream>>>(
      X, Wba, ba, ROWS, 64, HIDDEN);
  conv_silu<<<((size_t)ROWS * CONV_DIM) / 256, 256, 0, stream>>>(qkvz, cw, mixed);
  gates_kernel<<<(ROWS * H_V) / 256, 256, 0, stream>>>(ba, Alg, dtb, beta, g);
  l2norm_qk<<<(ROWS * 32) / 8, 256, 0, stream>>>(mixed);
  const size_t core_lds =
      (size_t)(128 * 128 + 5 * 64 * 128 + 2 * 64 * 64 + 128) * sizeof(float);
  gdn_core<<<BATCH * H_V, 256, core_lds, stream>>>(mixed, beta, g, core);
  rmsnorm_gate<<<(ROWS * H_V) / 8, 256, 0, stream>>>(core, qkvz, nw);
  gemm_bf16_wmma<<<dim3(HIDDEN / GN, ROWS / GM), 256, gemm_lds, stream>>>(
      core, Wo, out, ROWS, HIDDEN, VALUE_DIM);
}
